// VectorQuantizer_18116172054712
// MI455X (gfx1250) — compile-verified, matrix operand order checked by reading, not run on hardware
//
#include <hip/hip_runtime.h>
#include <hip/hip_bf16.h>

// ---------------------------------------------------------------------------
// VQ-VAE vector quantizer for MI455X (gfx1250, wave32, WMMA).
//
//   x:        [8,1024,1024] f32  -> rows: N=131072, D=64
//   codebook: [K=1024, D=64] f32
//   out = concat(loss[1], quantized[8388608], perplexity[1], idx[131072]) f32
//
// Strategy:
//   argmin_j ||x-e_j||^2  ==  argmax_j ( x.e_j - 0.5*||e_j||^2 )
//   dot computed with compensated bf16 split (xhi*ehi + xhi*elo + xlo*ehi)
//   via V_WMMA_F32_16X16X32_BF16, -0.5*||e||^2 (exact f32) as the C init.
//   Full codebook (hi+lo bf16, padded) lives in LDS (CDNA5: 320KB/WGP).
// ---------------------------------------------------------------------------

typedef __bf16 bf16_t;
typedef float  v4f  __attribute__((ext_vector_type(4)));
typedef float  v8f  __attribute__((ext_vector_type(8)));
typedef __bf16 v8bf __attribute__((ext_vector_type(8)));
typedef __bf16 v16bf __attribute__((ext_vector_type(16)));

#define VQ_N      131072          // rows of x (8*1024*1024 / 64)
#define VQ_K      1024            // codebook entries
#define VQ_D      64              // embedding dim
#define VQ_NUMEL  8388608.0f      // elements of x
#define LDS_STRIDE 72             // padded bf16 row stride: 36 dwords -> conflict-free

#define WMMA_BF16(A, B, C) \
    __builtin_amdgcn_wmma_f32_16x16x32_bf16(false, (A), false, (B), (short)0, (C), false, false)

// workspace layout (bytes)
#define WS_EHI   0                              // 1024*72*2 = 147456
#define WS_ELO   147456                         // 147456
#define WS_E2    294912                         // 1024*4 = 4096
#define WS_CNT   299008                         // 1024*4 = 4096
#define WS_LOSS  303104                         // 4
#define LDS_BYTES 299008                        // ehi + elo + e2 in LDS

__device__ __forceinline__ v16bf cat8(v8bf a, v8bf b) {
    return __builtin_shufflevector(a, b, 0,1,2,3,4,5,6,7,8,9,10,11,12,13,14,15);
}

// load 8 contiguous f32, produce bf16 hi + residual lo
__device__ __forceinline__ void load_cvt8(const float* p, v8bf& hi, v8bf& lo) {
    const v4f* q = (const v4f*)p;
    v4f a = q[0], b = q[1];
#pragma unroll
    for (int i = 0; i < 4; ++i) {
        float va = a[i];
        __bf16 ha = (__bf16)va;
        hi[i] = ha; lo[i] = (__bf16)(va - (float)ha);
        float vb = b[i];
        __bf16 hb = (__bf16)vb;
        hi[i + 4] = hb; lo[i + 4] = (__bf16)(vb - (float)hb);
    }
}

// --------------------------- prep kernel -----------------------------------
// codebook f32 -> bf16 hi/lo (padded rows) in ws, exact ||e||^2, zero counts.
__global__ void vq_prep(const float* __restrict__ cb,
                        bf16_t* __restrict__ ehi, bf16_t* __restrict__ elo,
                        float* __restrict__ e2, unsigned* __restrict__ counts,
                        float* __restrict__ loss_acc) {
    int tid = blockIdx.x * blockDim.x + threadIdx.x;
    int T = gridDim.x * blockDim.x;
    for (int i = tid; i < VQ_K * VQ_D; i += T) {
        int j = i >> 6, d = i & 63;
        float v = cb[i];
        __bf16 h = (__bf16)v;
        ehi[j * LDS_STRIDE + d] = h;
        elo[j * LDS_STRIDE + d] = (__bf16)(v - (float)h);
    }
    for (int j = tid; j < VQ_K; j += T) {
        float s = 0.0f;
#pragma unroll 8
        for (int d = 0; d < VQ_D; ++d) { float v = cb[j * VQ_D + d]; s += v * v; }
        e2[j] = s;
        counts[j] = 0u;
    }
    if (tid == 0) *loss_acc = 0.0f;
}

// --------------------------- main kernel -----------------------------------
__global__ __launch_bounds__(256) void vq_main(
        const float* __restrict__ x, const float* __restrict__ cb,
        const bf16_t* __restrict__ ehi_g, const bf16_t* __restrict__ elo_g,
        const float* __restrict__ e2_g,
        unsigned* __restrict__ counts, float* __restrict__ loss_acc,
        float* __restrict__ q_out, float* __restrict__ idx_out) {
    extern __shared__ char smem[];
    bf16_t* ehi_s = (bf16_t*)smem;                          // 147456 B
    bf16_t* elo_s = (bf16_t*)(smem + WS_ELO);               // 147456 B
    float*  e2_s  = (float*)(smem + WS_E2);                 // 4096 B

    // stage codebook (hi/lo bf16, padded) + ||e||^2 into LDS
    {
        const uint4* s0 = (const uint4*)ehi_g;
        uint4* d0 = (uint4*)ehi_s;
        const uint4* s1 = (const uint4*)elo_g;
        uint4* d1 = (uint4*)elo_s;
        const int n16 = (VQ_K * LDS_STRIDE * 2) / 16;       // 9216
        for (int i = threadIdx.x; i < n16; i += blockDim.x) { d0[i] = s0[i]; d1[i] = s1[i]; }
        for (int i = threadIdx.x; i < VQ_K; i += blockDim.x) e2_s[i] = e2_g[i];
    }
    __syncthreads();

    const int wave  = threadIdx.x >> 5;
    const int lane  = threadIdx.x & 31;
    const int lhalf = lane & 15;
    const bool upper = (lane >= 16);
    const int dlane = upper ? 8 : 0;                        // WMMA 16-bit K-offset per lane half

    float lsum = 0.0f;

    for (int pass = 0; pass < 4; ++pass) {
        const int chunk = blockIdx.x * 32 + wave * 4 + pass;   // 0..4095
        const int row_base = chunk * 32;                        // two 16-row M tiles

        // ---- load A fragments (x rows) with hi/lo bf16 split ----
        v16bf axh[2][2], axl[2][2];                             // [tile][k-half]
#pragma unroll
        for (int t = 0; t < 2; ++t) {
            const float* xr = x + (size_t)(row_base + t * 16 + lhalf) * VQ_D;
#pragma unroll
            for (int h = 0; h < 2; ++h) {
                int d0 = h * 32 + dlane;
                v8bf h0, l0, h1, l1;
                load_cvt8(xr + d0,      h0, l0);
                load_cvt8(xr + d0 + 16, h1, l1);
                axh[t][h] = cat8(h0, h1);
                axl[t][h] = cat8(l0, l1);
            }
        }

        float bval[2][8];
        int   bidx[2][8];
#pragma unroll
        for (int t = 0; t < 2; ++t)
#pragma unroll
            for (int r = 0; r < 8; ++r) { bval[t][r] = -3.402823466e38f; bidx[t][r] = 0; }

        // ---- sweep all 64 code tiles ----
        for (int jt = 0; jt < 64; ++jt) {
            const int j = jt * 16 + lhalf;                      // this lane's code column
            const float cinit = -0.5f * e2_s[j];
            const int jrow = j * LDS_STRIDE;

            v16bf bh[2], bl[2];
#pragma unroll
            for (int h = 0; h < 2; ++h) {
                int d0 = h * 32 + dlane;
                v8bf c0 = *(const v8bf*)(ehi_s + jrow + d0);
                v8bf c1 = *(const v8bf*)(ehi_s + jrow + d0 + 16);
                bh[h] = cat8(c0, c1);
                v8bf c2 = *(const v8bf*)(elo_s + jrow + d0);
                v8bf c3 = *(const v8bf*)(elo_s + jrow + d0 + 16);
                bl[h] = cat8(c2, c3);
            }

#pragma unroll
            for (int t = 0; t < 2; ++t) {
                v8f acc = {cinit, cinit, cinit, cinit, cinit, cinit, cinit, cinit};
                // compensated product: xhi*ehi + xhi*elo + xlo*ehi (f32 accum)
                acc = WMMA_BF16(axh[t][0], bh[0], acc);
                acc = WMMA_BF16(axh[t][1], bh[1], acc);
                acc = WMMA_BF16(axh[t][0], bl[0], acc);
                acc = WMMA_BF16(axh[t][1], bl[1], acc);
                acc = WMMA_BF16(axl[t][0], bh[0], acc);
                acc = WMMA_BF16(axl[t][1], bh[1], acc);
#pragma unroll
                for (int r = 0; r < 8; ++r) {
                    float v = acc[r];
                    if (v > bval[t][r]) { bval[t][r] = v; bidx[t][r] = j; }
                }
            }
        }

        // ---- cross-lane argmax butterfly within each 16-lane half ----
#pragma unroll
        for (int t = 0; t < 2; ++t) {
#pragma unroll
            for (int r = 0; r < 8; ++r) {
                float v = bval[t][r];
                int   i = bidx[t][r];
#pragma unroll
                for (int m = 8; m >= 1; m >>= 1) {
                    float v2 = __shfl_xor(v, m, 32);
                    int   i2 = __shfl_xor(i, m, 32);
                    if (v2 > v || (v2 == v && i2 < i)) { v = v2; i = i2; }
                }
                bidx[t][r] = i;     // all lanes in the half agree now
            }
        }

        // ---- gather exact f32 codes, write q & idx, accumulate exact loss ----
#pragma unroll
        for (int t = 0; t < 2; ++t) {
            const int rb = row_base + t * 16 + (upper ? 8 : 0);
#pragma unroll
            for (int m = 0; m < 8; ++m) {
                const int row = rb + m;
                const int ci = bidx[t][m];
                const v4f* cbp = (const v4f*)(cb + (size_t)ci * VQ_D);
                const v4f* xp  = (const v4f*)(x + (size_t)row * VQ_D);
                v4f q = cbp[lhalf];
                v4f xv = xp[lhalf];
                float* qo = q_out + (size_t)row * VQ_D + lhalf * 4;
#pragma unroll
                for (int c = 0; c < 4; ++c) {
                    qo[c] = q[c];                         // d_out+1 base: keep 4B stores
                    float d = q[c] - xv[c];
                    lsum += d * d;
                }
                if (lhalf == m) {
                    idx_out[row] = (float)ci;
                    atomicAdd(&counts[ci], 1u);
                }
            }
        }
    }

    // wave-reduce loss, one atomic per wave
#pragma unroll
    for (int m = 16; m >= 1; m >>= 1) lsum += __shfl_xor(lsum, m, 32);
    if (lane == 0) atomicAdd(loss_acc, lsum);
}

// --------------------------- finalize kernel -------------------------------
__global__ void vq_final(const unsigned* __restrict__ counts,
                         const float* __restrict__ loss_acc,
                         float* __restrict__ out_loss, float* __restrict__ out_ppl) {
    __shared__ float sh[256];
    float h = 0.0f;
    for (int j = threadIdx.x; j < VQ_K; j += 256) {
        float p = (float)counts[j] * (1.0f / (float)VQ_N);
        h -= p * logf(p + 1e-10f);
    }
    sh[threadIdx.x] = h;
    __syncthreads();
    for (int s = 128; s > 0; s >>= 1) {
        if (threadIdx.x < s) sh[threadIdx.x] += sh[threadIdx.x + s];
        __syncthreads();
    }
    if (threadIdx.x == 0) {
        *out_ppl = expf(sh[0]);
        // loss = q_latent + 0.25*e_latent = 1.25 * mean((q-x)^2)
        *out_loss = (*loss_acc) * (1.25f / VQ_NUMEL);
    }
}

// ---------------------------------------------------------------------------
extern "C" void kernel_launch(void* const* d_in, const int* in_sizes, int n_in,
                              void* d_out, int out_size, void* d_ws, size_t ws_size,
                              hipStream_t stream) {
    (void)in_sizes; (void)n_in; (void)out_size; (void)ws_size;
    const float* x  = (const float*)d_in[0];
    const float* cb = (const float*)d_in[1];

    char* ws = (char*)d_ws;
    bf16_t*   ehi    = (bf16_t*)(ws + WS_EHI);
    bf16_t*   elo    = (bf16_t*)(ws + WS_ELO);
    float*    e2     = (float*)(ws + WS_E2);
    unsigned* counts = (unsigned*)(ws + WS_CNT);
    float*    lacc   = (float*)(ws + WS_LOSS);

    float* out      = (float*)d_out;
    float* out_loss = out;                        // [0]
    float* out_q    = out + 1;                    // [1 .. 8388608]
    float* out_ppl  = out + 1 + 8388608;          // [8388609]
    float* out_idx  = out + 2 + 8388608;          // [8388610 ..]

    // allow ~292 KB dynamic LDS (CDNA5: 320 KB per WGP)
    (void)hipFuncSetAttribute((const void*)vq_main,
                              hipFuncAttributeMaxDynamicSharedMemorySize, LDS_BYTES);

    vq_prep<<<16, 256, 0, stream>>>(cb, ehi, elo, e2, counts, lacc);
    vq_main<<<128, 256, LDS_BYTES, stream>>>(x, cb, ehi, elo, e2, counts, lacc,
                                             out_q, out_idx);
    vq_final<<<1, 256, 0, stream>>>(counts, lacc, out_loss, out_ppl);
}
